// MambaProj_8804682957495
// MI455X (gfx1250) — matrix operand order repaired and checked by reading, not compiled, not run on hardware
//
#include <hip/hip_runtime.h>
#include <hip/hip_bf16.h>
#include <stdint.h>

// ---------------------------------------------------------------------------
// Mamba (bidirectional) block for MI455X / gfx1250, wave32 + WMMA bf16x3.
// Operands pre-packed as bf16 hi/lo in the WMMA fragment layout (each
// fragment = 2x global_load_b128). Big GEMMs use 2x2 register blocking:
// 12 WMMAs per 16 b128 loads per K=32 chunk.
// ---------------------------------------------------------------------------

#define D_MODEL 512
#define D_INNER 512
#define D_HALF  256
#define D_STATE 8
#define DT_RANK 32
#define BSZ     2
#define LEN     4096
#define NSTREAM 4                          // 2 batches x {fwd, bwd}
#define XDBL_E  (DT_RANK + 2 * D_STATE)    // 48
#define KOUT    (2 * D_INNER)              // 1024 (out_proj K)

typedef __attribute__((ext_vector_type(16))) __bf16 v16bf;
typedef __attribute__((ext_vector_type(8)))  float  v8f;
typedef unsigned short u16;

union Frag { v16bf v; uint4 q[2]; };

__device__ __forceinline__ u16 f2bf(float f) {
  unsigned u = __float_as_uint(f);
  u += 0x7FFFu + ((u >> 16) & 1u);          // round-to-nearest-even
  return (u16)(u >> 16);
}
__device__ __forceinline__ float bf2f(u16 b) {
  return __uint_as_float(((unsigned)b) << 16);
}

// A fragment: row-major bf16 [m][K], lane = 16*kh + m.
// K indices: {kh*8 .. kh*8+7} and {16+kh*8 .. 16+kh*8+7}  -> two b128 loads.
__device__ __forceinline__ void ldA(Frag& f, const u16* rowbase, int kh) {
  f.q[0] = *reinterpret_cast<const uint4*>(rowbase + kh * 8);
  f.q[1] = *reinterpret_cast<const uint4*>(rowbase + kh * 8 + 16);
}
// B fragment: row-major bf16 [n][K], lane = 16*kh + n.
// K indices: {kh*8 .. kh*8+7} and {16+kh*8 .. 16+kh*8+7}  -> two b128 loads.
__device__ __forceinline__ void ldB(Frag& f, const u16* rowbase, int kh) {
  f.q[0] = *reinterpret_cast<const uint4*>(rowbase + kh * 8);
  f.q[1] = *reinterpret_cast<const uint4*>(rowbase + kh * 8 + 16);
}

__device__ __forceinline__ v8f wmma_bf16x3(const Frag& ah, const Frag& al,
                                           const Frag& bh, const Frag& bl,
                                           v8f acc) {
  acc = __builtin_amdgcn_wmma_f32_16x16x32_bf16(false, ah.v, false, bh.v,
                                                (short)0, acc, false, false);
  acc = __builtin_amdgcn_wmma_f32_16x16x32_bf16(false, al.v, false, bh.v,
                                                (short)0, acc, false, false);
  acc = __builtin_amdgcn_wmma_f32_16x16x32_bf16(false, ah.v, false, bl.v,
                                                (short)0, acc, false, false);
  return acc;
}

// 2x2-blocked bf16x3 GEMM core: wave computes a 32x32 tile of A(MxK)*B(NxK)^T.
template <int K>
__device__ __forceinline__ void gemm_2x2(
    const u16* __restrict__ ah_, const u16* __restrict__ al_,
    const u16* __restrict__ bh_, const u16* __restrict__ bl_,
    int row0, int col0, int m, int kh, v8f acc[2][2]) {
  const u16* arh0 = ah_ + (size_t)(row0 + m) * K;
  const u16* arh1 = arh0 + 16 * K;
  const u16* arl0 = al_ + (size_t)(row0 + m) * K;
  const u16* arl1 = arl0 + 16 * K;
  const u16* brh0 = bh_ + (size_t)(col0 + m) * K;
  const u16* brh1 = brh0 + 16 * K;
  const u16* brl0 = bl_ + (size_t)(col0 + m) * K;
  const u16* brl1 = brl0 + 16 * K;
  for (int k0 = 0; k0 < K; k0 += 32) {
    __builtin_prefetch(arh0 + k0 + 128, 0, 3);
    __builtin_prefetch(brh0 + k0 + 128, 0, 3);
    Frag a0h, a0l, a1h, a1l, b0h, b0l, b1h, b1l;
    ldA(a0h, arh0 + k0, kh); ldA(a0l, arl0 + k0, kh);
    ldA(a1h, arh1 + k0, kh); ldA(a1l, arl1 + k0, kh);
    ldB(b0h, brh0 + k0, kh); ldB(b0l, brl0 + k0, kh);
    ldB(b1h, brh1 + k0, kh); ldB(b1l, brl1 + k0, kh);
    acc[0][0] = wmma_bf16x3(a0h, a0l, b0h, b0l, acc[0][0]);
    acc[0][1] = wmma_bf16x3(a0h, a0l, b1h, b1l, acc[0][1]);
    acc[1][0] = wmma_bf16x3(a1h, a1l, b0h, b0l, acc[1][0]);
    acc[1][1] = wmma_bf16x3(a1h, a1l, b1h, b1l, acc[1][1]);
  }
}

// ---------------------------------------------------------------------------
// P0: generic fp32 -> bf16 hi/lo packer (runs once per operand element)
// ---------------------------------------------------------------------------
__global__ void k_pack(const float* __restrict__ in, u16* __restrict__ oh,
                       u16* __restrict__ ol, int n) {
  int i = blockIdx.x * blockDim.x + threadIdx.x;
  if (i >= n) return;
  float f = in[i];
  u16 h = f2bf(f);
  oh[i] = h;
  ol[i] = f2bf(f - bf2f(h));
}

// ---------------------------------------------------------------------------
// K1: in_proj GEMM  xz[b,e,l] = sum_d hs[b,l,d] * W[e,d]
//     A = packed hs (8192 x 512), B = packed W (1024 x 512). 32x32 per wave.
//     Fused stream rearrangement (+ time reversal) on store.
// ---------------------------------------------------------------------------
__global__ __launch_bounds__(128) void k_inproj(
    const u16* __restrict__ ah_, const u16* __restrict__ al_,
    const u16* __restrict__ bh_, const u16* __restrict__ bl_,
    float* __restrict__ xraw, float* __restrict__ zraw) {
  const int NT = KOUT / 32;                                // 32 col tiles
  int gt   = blockIdx.x * (blockDim.x >> 5) + (threadIdx.x >> 5);
  int lane = threadIdx.x & 31;
  int mt = gt / NT, nt = gt % NT;
  int row0 = mt * 32, col0 = nt * 32;
  int m = lane & 15, kh = lane >> 4;

  v8f acc[2][2] = {};
  gemm_2x2<D_MODEL>(ah_, al_, bh_, bl_, row0, col0, m, kh, acc);

#pragma unroll
  for (int ct = 0; ct < 2; ++ct) {
    int e    = col0 + ct * 16 + (lane & 15);
    int quad = e >> 8;               // 0:x fwd 1:z fwd 2:x bwd 3:z bwd
    int ch   = e & 255;
    float* dst = (quad & 1) ? zraw : xraw;
#pragma unroll
    for (int rt = 0; rt < 2; ++rt) {
#pragma unroll
      for (int r = 0; r < 8; ++r) {
        int mm = row0 + rt * 16 + r + 8 * kh;
        int b = mm >> 12, l = mm & (LEN - 1);
        int s  = (quad >> 1) * 2 + b;
        int ll = (quad >= 2) ? (LEN - 1 - l) : l;
        dst[((size_t)s * D_HALF + ch) * LEN + ll] = acc[rt][ct][r];
      }
    }
  }
}

// ---------------------------------------------------------------------------
// K2: depthwise causal conv (x: dil 8, z: dil 1) + SiLU.
//     Emits: xact fp32 [s][d][l]          (scan input u)
//            xactT bf16 hi/lo [s][l][d]   (B operand of x_proj GEMM)
//            z -> Aout bf16 hi/lo, gathered layout for out_proj A
// ---------------------------------------------------------------------------
__global__ void k_conv(const float* __restrict__ xraw, const float* __restrict__ zraw,
                       const float* __restrict__ wx, const float* __restrict__ wz,
                       float* __restrict__ xact,
                       u16* __restrict__ xactT_h, u16* __restrict__ xactT_l,
                       u16* __restrict__ Aout_h,  u16* __restrict__ Aout_l) {
  int idx = blockIdx.x * blockDim.x + threadIdx.x;
  if (idx >= NSTREAM * D_HALF * LEN) return;
  int l = idx & (LEN - 1);
  int d = (idx >> 12) & (D_HALF - 1);
  int s = idx >> 20;
  float ax = 0.f, az = 0.f;
#pragma unroll
  for (int j = 0; j < 4; ++j) {
    int lx = l - (3 - j) * 8;
    if (lx >= 0) ax += xraw[idx + (lx - l)] * wx[d * 4 + j];
    int lz = l - (3 - j);
    if (lz >= 0) az += zraw[idx + (lz - l)] * wz[d * 4 + j];
  }
  float sx = ax / (1.f + __expf(-ax));      // silu
  float sz = az / (1.f + __expf(-az));
  xact[idx] = sx;
  size_t t = ((size_t)s * LEN + l) * D_HALF + d;
  u16 hx = f2bf(sx);
  xactT_h[t] = hx;
  xactT_l[t] = f2bf(sx - bf2f(hx));
  // z contribution to out_proj A matrix (c = 256+d fwd, 768+d bwd, l reversed)
  size_t row = (s < 2) ? ((size_t)s * LEN + l)
                       : ((size_t)(s - 2) * LEN + (LEN - 1 - l));
  int col = ((s < 2) ? 256 : 768) + d;
  u16 hz = f2bf(sz);
  Aout_h[row * KOUT + col] = hz;
  Aout_l[row * KOUT + col] = f2bf(sz - bf2f(hz));
}

// ---------------------------------------------------------------------------
// K3: x_dbl GEMM per stream: xdbl[s,e,l] = sum_d xw[e,d] * xact[s,d,l]
//     A = packed x_proj_w (48 x 256), B = xactT (l-major). M=48,N=4096,K=256.
//     Also emits dtrT bf16 hi/lo [s][l][e<32] (B operand of dt GEMM).
// ---------------------------------------------------------------------------
__global__ __launch_bounds__(128) void k_xproj(
    const u16* __restrict__ ah_, const u16* __restrict__ al_,
    const u16* __restrict__ xactT_h, const u16* __restrict__ xactT_l,
    float* __restrict__ xdbl,
    u16* __restrict__ dtrT_h, u16* __restrict__ dtrT_l) {
  const int MT = XDBL_E / 16, LT = LEN / 16;               // 3, 256
  int gt   = blockIdx.x * (blockDim.x >> 5) + (threadIdx.x >> 5);
  int lane = threadIdx.x & 31;
  int s = gt / (MT * LT);
  int mt = (gt / LT) % MT;
  int l0 = (gt % LT) * 16;
  int m = lane & 15, kh = lane >> 4;

  const u16* arh = ah_ + (size_t)(mt * 16 + m) * D_HALF;
  const u16* arl = al_ + (size_t)(mt * 16 + m) * D_HALF;
  const u16* brh = xactT_h + ((size_t)s * LEN + l0 + m) * D_HALF;
  const u16* brl = xactT_l + ((size_t)s * LEN + l0 + m) * D_HALF;

  v8f acc = {};
  for (int k0 = 0; k0 < D_HALF; k0 += 32) {
    Frag ah, al, bh, bl;
    ldA(ah, arh + k0, kh); ldA(al, arl + k0, kh);
    ldB(bh, brh + k0, kh); ldB(bl, brl + k0, kh);
    acc = wmma_bf16x3(ah, al, bh, bl, acc);
  }
  int l = l0 + (lane & 15);
#pragma unroll
  for (int r = 0; r < 8; ++r) {
    int e = mt * 16 + r + 8 * kh;
    xdbl[((size_t)s * XDBL_E + e) * LEN + l] = acc[r];
  }
  if (mt < 2) {                       // e < 32: dtr rows, pack for dt GEMM
#pragma unroll
    for (int r = 0; r < 8; ++r) {
      int e = mt * 16 + r + 8 * kh;
      size_t t = ((size_t)s * LEN + l) * DT_RANK + e;
      u16 h = f2bf(acc[r]);
      dtrT_h[t] = h;
      dtrT_l[t] = f2bf(acc[r] - bf2f(h));
    }
  }
}

// ---------------------------------------------------------------------------
// K4: dt GEMM + softplus. A = packed dt_proj_w (256 x 32), B = dtrT.
//     M=256, N=4096, K=32 (single WMMA chunk). delta[s,d,l] fp32 out.
// ---------------------------------------------------------------------------
__global__ __launch_bounds__(128) void k_dt(
    const u16* __restrict__ ah_, const u16* __restrict__ al_,
    const float* __restrict__ dtb,
    const u16* __restrict__ dtrT_h, const u16* __restrict__ dtrT_l,
    float* __restrict__ delta) {
  const int MT = D_HALF / 16, LT = LEN / 16;               // 16, 256
  int gt   = blockIdx.x * (blockDim.x >> 5) + (threadIdx.x >> 5);
  int lane = threadIdx.x & 31;
  int s = gt / (MT * LT);
  int mt = (gt / LT) % MT;
  int l0 = (gt % LT) * 16;
  int m = lane & 15, kh = lane >> 4;

  Frag ah, al, bh, bl;
  ldA(ah, ah_ + (size_t)(mt * 16 + m) * DT_RANK, kh);
  ldA(al, al_ + (size_t)(mt * 16 + m) * DT_RANK, kh);
  ldB(bh, dtrT_h + ((size_t)s * LEN + l0 + m) * DT_RANK, kh);
  ldB(bl, dtrT_l + ((size_t)s * LEN + l0 + m) * DT_RANK, kh);
  v8f acc = {};
  acc = wmma_bf16x3(ah, al, bh, bl, acc);

  int l = l0 + (lane & 15);
#pragma unroll
  for (int r = 0; r < 8; ++r) {
    int d = mt * 16 + r + 8 * kh;
    float v = acc[r] + dtb[d];
    float sp = (v > 20.f) ? v : log1pf(__expf(v));         // softplus
    delta[((size_t)s * D_HALF + d) * LEN + l] = sp;
  }
}

// ---------------------------------------------------------------------------
// K5: selective scan. 1 block/stream, 1 thread/channel, 8 states in regs,
//     B & C staged in LDS. Writes y straight into the out_proj A matrix
//     (bf16 hi/lo, gathered layout, time reversal folded in).
// ---------------------------------------------------------------------------
#define SCAN_CHUNK 128
__global__ __launch_bounds__(256) void k_scan(
    const float* __restrict__ xact, const float* __restrict__ xdbl,
    const float* __restrict__ delta, const float* __restrict__ A_log,
    const float* __restrict__ Dp,
    u16* __restrict__ Aout_h, u16* __restrict__ Aout_l) {
  int s = blockIdx.x;
  int d = threadIdx.x;
  float Av[D_STATE], h[D_STATE];
#pragma unroll
  for (int n = 0; n < D_STATE; ++n) {
    Av[n] = -__expf(A_log[d * D_STATE + n]);
    h[n] = 0.f;
  }
  float Dd = Dp[d];
  __shared__ float sB[D_STATE][SCAN_CHUNK];
  __shared__ float sC[D_STATE][SCAN_CHUNK];
  const float* Bb = xdbl + ((size_t)s * XDBL_E + DT_RANK) * LEN;
  const float* Cb = xdbl + ((size_t)s * XDBL_E + DT_RANK + D_STATE) * LEN;
  const float* up = xact  + ((size_t)s * D_HALF + d) * LEN;
  const float* dp = delta + ((size_t)s * D_HALF + d) * LEN;
  int col = (s < 2) ? d : (512 + d);
  int sb  = (s < 2) ? s : (s - 2);

  for (int l0 = 0; l0 < LEN; l0 += SCAN_CHUNK) {
    __syncthreads();
    for (int t = threadIdx.x; t < D_STATE * SCAN_CHUNK; t += blockDim.x) {
      int n = t / SCAN_CHUNK, j = t % SCAN_CHUNK;
      sB[n][j] = Bb[(size_t)n * LEN + l0 + j];
      sC[n][j] = Cb[(size_t)n * LEN + l0 + j];
    }
    __syncthreads();
    for (int j = 0; j < SCAN_CHUNK; ++j) {
      float u = up[l0 + j], de = dp[l0 + j];
      float accv = u * Dd;
      float dbu = de * u;
#pragma unroll
      for (int n = 0; n < D_STATE; ++n) {
        float dA = __expf(de * Av[n]);
        h[n] = dA * h[n] + dbu * sB[n][j];
        accv += h[n] * sC[n][j];
      }
      int l = l0 + j;
      size_t row = (size_t)sb * LEN + ((s < 2) ? l : (LEN - 1 - l));
      u16 hb = f2bf(accv);
      Aout_h[row * KOUT + col] = hb;
      Aout_l[row * KOUT + col] = f2bf(accv - bf2f(hb));
    }
  }
}

// ---------------------------------------------------------------------------
// K6: out_proj GEMM: out[b,l,o] = sum_c A[b*L+l][c] * Wout[o,c]
//     A = pre-gathered bf16 hi/lo (8192 x 1024), B = packed Wout (512 x 1024).
//     32x32 per wave.
// ---------------------------------------------------------------------------
__global__ __launch_bounds__(128) void k_outproj(
    const u16* __restrict__ ah_, const u16* __restrict__ al_,
    const u16* __restrict__ bh_, const u16* __restrict__ bl_,
    float* __restrict__ out) {
  const int NT = D_MODEL / 32;                             // 16 col tiles
  int gt   = blockIdx.x * (blockDim.x >> 5) + (threadIdx.x >> 5);
  int lane = threadIdx.x & 31;
  int mt = gt / NT, nt = gt % NT;
  int row0 = mt * 32, col0 = nt * 32;
  int m = lane & 15, kh = lane >> 4;

  v8f acc[2][2] = {};
  gemm_2x2<KOUT>(ah_, al_, bh_, bl_, row0, col0, m, kh, acc);

  int n = lane & 15;
#pragma unroll
  for (int ct = 0; ct < 2; ++ct) {
#pragma unroll
    for (int rt = 0; rt < 2; ++rt) {
#pragma unroll
      for (int r = 0; r < 8; ++r) {
        int mr = row0 + rt * 16 + r + 8 * kh;
        int bb = mr >> 12, lr = mr & (LEN - 1);
        out[((size_t)bb * LEN + lr) * D_MODEL + col0 + ct * 16 + n] = acc[rt][ct][r];
      }
    }
  }
}

// ---------------------------------------------------------------------------
// Host launch
// ---------------------------------------------------------------------------
extern "C" void kernel_launch(void* const* d_in, const int* in_sizes, int n_in,
                              void* d_out, int out_size, void* d_ws, size_t ws_size,
                              hipStream_t stream) {
  (void)in_sizes; (void)n_in; (void)out_size; (void)ws_size;
  const float* hs   = (const float*)d_in[0];
  const float* Win  = (const float*)d_in[1];
  const float* xw   = (const float*)d_in[2];
  const float* dtw  = (const float*)d_in[3];
  const float* dtb  = (const float*)d_in[4];
  const float* Alog = (const float*)d_in[5];
  const float* Dprm = (const float*)d_in[6];
  const float* wx   = (const float*)d_in[7];
  const float* wz   = (const float*)d_in[8];
  const float* Wout = (const float*)d_in[9];
  float* out = (float*)d_out;

  // ---- workspace layout ----------------------------------------------------
  const size_t NBIG = (size_t)NSTREAM * D_HALF * LEN;      // 4,194,304
  float* f32b = (float*)d_ws;
  float* xraw = f32b;                       // NBIG
  float* zraw = xraw + NBIG;                // NBIG
  float* xact = zraw + NBIG;                // NBIG
  float* xdbl = xact + NBIG;                // 4*48*4096 = 786,432
  u16* ub      = (u16*)(xdbl + (size_t)NSTREAM * XDBL_E * LEN);
  u16* hsA_h   = ub;                         size_t n_hs   = (size_t)BSZ * LEN * D_MODEL;   // 4,194,304
  u16* hsA_l   = hsA_h   + n_hs;
  u16* WinP_h  = hsA_l   + n_hs;             size_t n_win  = (size_t)KOUT * D_MODEL;        // 524,288
  u16* WinP_l  = WinP_h  + n_win;
  u16* xwP_h   = WinP_l  + n_win;            size_t n_xw   = (size_t)XDBL_E * D_HALF;       // 12,288
  u16* xwP_l   = xwP_h   + n_xw;
  u16* dtwP_h  = xwP_l   + n_xw;             size_t n_dtw  = (size_t)D_HALF * DT_RANK;      // 8,192
  u16* dtwP_l  = dtwP_h  + n_dtw;
  u16* WoutP_h = dtwP_l  + n_dtw;            size_t n_wout = (size_t)D_MODEL * KOUT;        // 524,288
  u16* WoutP_l = WoutP_h + n_wout;
  u16* xactT_h = WoutP_l + n_wout;           // NBIG
  u16* xactT_l = xactT_h + NBIG;
  u16* dtrT_h  = xactT_l + NBIG;             size_t n_dtr  = (size_t)NSTREAM * LEN * DT_RANK; // 524,288
  u16* dtrT_l  = dtrT_h  + n_dtr;
  u16* Aout_h  = dtrT_l  + n_dtr;            size_t n_aout = (size_t)BSZ * LEN * KOUT;      // 8,388,608
  u16* Aout_l  = Aout_h  + n_aout;
  // hs pack region is dead after K1 -> reuse as fp32 delta (exact size match)
  float* delta = (float*)hsA_h;              // NBIG floats

  const int WPB = 4;                         // waves (tiles) per 128-thread block

  // ---- P0: pack static operands (bf16 hi/lo) -------------------------------
  k_pack<<<(int)((n_hs   + 255) / 256), 256, 0, stream>>>(hs,   hsA_h,  hsA_l,  (int)n_hs);
  k_pack<<<(int)((n_win  + 255) / 256), 256, 0, stream>>>(Win,  WinP_h, WinP_l, (int)n_win);
  k_pack<<<(int)((n_xw   + 255) / 256), 256, 0, stream>>>(xw,   xwP_h,  xwP_l,  (int)n_xw);
  k_pack<<<(int)((n_dtw  + 255) / 256), 256, 0, stream>>>(dtw,  dtwP_h, dtwP_l, (int)n_dtw);
  k_pack<<<(int)((n_wout + 255) / 256), 256, 0, stream>>>(Wout, WoutP_h, WoutP_l, (int)n_wout);

  // ---- K1: in_proj (256 x 32 wave tiles of 32x32) -------------------------
  {
    int tiles = (BSZ * LEN / 32) * (KOUT / 32);
    k_inproj<<<tiles / WPB, 32 * WPB, 0, stream>>>(hsA_h, hsA_l, WinP_h, WinP_l,
                                                   xraw, zraw);
  }
  // ---- K2: conv + silu + operand packing ----------------------------------
  {
    int n = NSTREAM * D_HALF * LEN;
    k_conv<<<(n + 255) / 256, 256, 0, stream>>>(xraw, zraw, wx, wz, xact,
                                                xactT_h, xactT_l, Aout_h, Aout_l);
  }
  // ---- K3: x_proj (4 x 3 x 256 tiles) -------------------------------------
  {
    int tiles = NSTREAM * (XDBL_E / 16) * (LEN / 16);
    k_xproj<<<tiles / WPB, 32 * WPB, 0, stream>>>(xwP_h, xwP_l, xactT_h, xactT_l,
                                                  xdbl, dtrT_h, dtrT_l);
  }
  // ---- K4: dt_proj + softplus (4 x 16 x 256 tiles) [delta overwrites hs pack]
  {
    int tiles = NSTREAM * (D_HALF / 16) * (LEN / 16);
    k_dt<<<tiles / WPB, 32 * WPB, 0, stream>>>(dtwP_h, dtwP_l, dtb,
                                               dtrT_h, dtrT_l, delta);
  }
  // ---- K5: selective scan (writes y into out_proj A matrix) ---------------
  k_scan<<<NSTREAM, D_HALF, 0, stream>>>(xact, xdbl, delta, Alog, Dprm,
                                         Aout_h, Aout_l);
  // ---- K6: out_proj (256 x 16 wave tiles of 32x32) ------------------------
  {
    int tiles = (BSZ * LEN / 32) * (D_MODEL / 32);
    k_outproj<<<tiles / WPB, 32 * WPB, 0, stream>>>(Aout_h, Aout_l,
                                                    WoutP_h, WoutP_l, out);
  }
}
